// TwoTowerMLP_7997229105388
// MI455X (gfx1250) — matrix operand order repaired and checked by reading, not compiled, not run on hardware
//
#include <hip/hip_runtime.h>
#include <hip/hip_bf16.h>
#include <math.h>

// ---------------------------------------------------------------------------
// Problem constants (from reference)
// ---------------------------------------------------------------------------
#define BATCH   32
#define LQ      64
#define LK      8192
#define DIM     256
#define H1      512
#define H2      512
#define NSPLIT  16
#define KSPLIT  (LK / NSPLIT)   // 512 keys per workgroup
#define CHUNK   64              // keys per inner chunk
#define NCHUNK  (KSPLIT / CHUNK)
#define NEGBIG  -1.0e30f
#define SCALE   0.0625f         // 1/sqrt(256)

typedef __attribute__((ext_vector_type(16))) __bf16 v16bf;
typedef __attribute__((ext_vector_type(8)))  float  v8f;
typedef __attribute__((ext_vector_type(4)))  unsigned int u32x4;

union BFPack { u32x4 q[2]; v16bf v; };

// A-matrix 16x32 bf16 layout (wave32):
//  lane<16: M=lane,    K elements {base..base+7, base+16..base+23} with base=0
//  lane>=16:M=lane-16, same with base=8
__device__ __forceinline__ v16bf load_tile_A(const __bf16* src, int row0, int ld,
                                             int cbase, int lane) {
  const __bf16* p = src + (row0 + (lane & 15)) * ld + cbase + ((lane < 16) ? 0 : 8);
  BFPack pk;
  pk.q[0] = *(const u32x4*)(p);        // K = base .. base+7
  pk.q[1] = *(const u32x4*)(p + 16);   // K = base+16 .. base+23
  return pk.v;
}

// B-matrix 32x16 bf16 layout (wave32): lane holds column n=lane&15,
// K elements {base..base+7, base+16..base+23} with base = (lane<16)?0:8.
// src is stored "column-major as rows": row n of src holds column n of B.
__device__ __forceinline__ v16bf load_tile_B(const __bf16* src, int col0, int ld,
                                             int cbase, int lane) {
  const __bf16* p = src + (col0 + (lane & 15)) * ld + cbase + ((lane < 16) ? 0 : 8);
  BFPack pk;
  pk.q[0] = *(const u32x4*)(p);
  pk.q[1] = *(const u32x4*)(p + 16);
  return pk.v;
}

__device__ __forceinline__ v8f wmma_bf16(v16bf a, v16bf b, v8f c) {
  return __builtin_amdgcn_wmma_f32_16x16x32_bf16(false, a, false, b, (short)0, c,
                                                 false, false);
}

// ---------------------------------------------------------------------------
// K0: convert Wq / Wkv (f32 [D,D] row-major) to bf16 TRANSPOSED layout:
//     WT[d*D + c] = W[c*D + d]   (so WMMA B-lanes read contiguous bytes)
// ---------------------------------------------------------------------------
__global__ __launch_bounds__(256) void k0_convert_weights(
    const float* __restrict__ Wq, const float* __restrict__ Wkv,
    __bf16* __restrict__ WqT, __bf16* __restrict__ WkvT) {
  int idx = blockIdx.x * 256 + threadIdx.x;          // 0 .. 2*D*D-1
  const float* src = (idx < DIM * DIM) ? Wq : Wkv;
  __bf16* dst      = (idx < DIM * DIM) ? WqT : WkvT;
  int j = (idx < DIM * DIM) ? idx : idx - DIM * DIM;
  int d = j >> 8, c = j & 255;
  dst[d * DIM + c] = (__bf16)src[c * DIM + d];
}

// ---------------------------------------------------------------------------
// K1: Q = miRNA_hidden @ Wq + bq  -> bf16 [B, LQ, D]
// one workgroup (8 waves) per batch row; 64 output tiles of 16x16, 8 per wave
// ---------------------------------------------------------------------------
__global__ __launch_bounds__(256) void k1_qproj(
    const float* __restrict__ mirna, const __bf16* __restrict__ WqT,
    const float* __restrict__ bq, __bf16* __restrict__ Qbf) {
  __shared__ __bf16 sXq[LQ * DIM];
  const int b = blockIdx.x, tid = threadIdx.x;
  const int lane = tid & 31, wave = tid >> 5;

  const float* xs = mirna + (size_t)b * LQ * DIM;
  for (int i = tid; i < LQ * DIM / 4; i += 256) {
    float4 v = reinterpret_cast<const float4*>(xs)[i];
    int o = i * 4;
    sXq[o + 0] = (__bf16)v.x; sXq[o + 1] = (__bf16)v.y;
    sXq[o + 2] = (__bf16)v.z; sXq[o + 3] = (__bf16)v.w;
  }
  __syncthreads();

  #pragma unroll
  for (int i = 0; i < 8; ++i) {
    int t = wave * 8 + i;
    int m0 = (t & 3) * 16, n0 = (t >> 2) * 16;
    v8f c = {};
    #pragma unroll
    for (int cb = 0; cb < DIM; cb += 32) {
      v16bf a  = load_tile_A(sXq, m0, DIM, cb, lane);
      v16bf bm = load_tile_B(WqT, n0, DIM, cb, lane);
      c = wmma_bf16(a, bm, c);
    }
    int n = n0 + (lane & 15);
    float bias = bq[n];
    int hi8 = (lane < 16) ? 0 : 8;
    #pragma unroll
    for (int r = 0; r < 8; ++r)
      Qbf[((size_t)b * LQ + m0 + r + hi8) * DIM + n] = (__bf16)(c[r] + bias);
  }
}

// ---------------------------------------------------------------------------
// K2: fused K-projection + flash attention over a 512-key slice.
// grid = B * NSPLIT workgroups, 256 threads (8 waves).
// Dynamic LDS (~153 KB; WGP has 320 KB):
//   sQ  [64x256 bf16]  sX [64x256 bf16]  sK [64x256 bf16]  sKT [256x64 bf16]
//   sS  [64x64 f32]    sP [64x64 bf16]   + row-state arrays
// ---------------------------------------------------------------------------
#define SM_SQ   0
#define SM_SX   (SM_SQ  + LQ * DIM * 2)
#define SM_SK   (SM_SX  + CHUNK * DIM * 2)
#define SM_SKT  (SM_SK  + CHUNK * DIM * 2)
#define SM_SS   (SM_SKT + DIM * CHUNK * 2)
#define SM_SP   (SM_SS  + LQ * CHUNK * 4)
#define SM_MASK (SM_SP  + LQ * CHUNK * 2)
#define SM_MROW (SM_MASK + CHUNK * 4)
#define SM_LROW (SM_MROW + LQ * 4)
#define SM_CORR (SM_LROW + LQ * 4)
#define SM_TOTAL (SM_CORR + LQ * 4)

__global__ __launch_bounds__(256) void k2_flash_attn(
    const float* __restrict__ mrna,      // [B, LK, D]
    const float* __restrict__ kmask,     // [B, LK]
    const __bf16* __restrict__ Qbf,      // [B, LQ, D]
    const __bf16* __restrict__ WkvT,     // [D, D] transposed bf16
    const float* __restrict__ bkv,       // [D]
    float* __restrict__ Mout,            // [B, NSPLIT, LQ]
    float* __restrict__ Lout,            // [B, NSPLIT, LQ]
    float* __restrict__ Acc)             // [B, NSPLIT, LQ, D]
{
  extern __shared__ char smem[];
  __bf16* sQ    = (__bf16*)(smem + SM_SQ);
  __bf16* sX    = (__bf16*)(smem + SM_SX);
  __bf16* sK    = (__bf16*)(smem + SM_SK);
  __bf16* sKT   = (__bf16*)(smem + SM_SKT);
  float*  sS    = (float*) (smem + SM_SS);
  __bf16* sP    = (__bf16*)(smem + SM_SP);
  float*  sMask = (float*) (smem + SM_MASK);
  float*  sMrow = (float*) (smem + SM_MROW);
  float*  sLrow = (float*) (smem + SM_LROW);
  float*  sCorr = (float*) (smem + SM_CORR);

  const int b     = blockIdx.x / NSPLIT;
  const int split = blockIdx.x % NSPLIT;
  const int tid   = threadIdx.x;
  const int lane  = tid & 31, wave = tid >> 5;
  const int hi8   = (lane < 16) ? 0 : 8;

  // stage Q into LDS (already bf16)
  {
    const u32x4* s4 = (const u32x4*)(Qbf + (size_t)b * LQ * DIM);
    u32x4* d4 = (u32x4*)sQ;
    for (int i = tid; i < LQ * DIM / 8; i += 256) d4[i] = s4[i];
  }
  if (tid < LQ) { sMrow[tid] = NEGBIG; sLrow[tid] = 0.f; }

  v8f acc[8];
  #pragma unroll
  for (int i = 0; i < 8; ++i) acc[i] = (v8f){};

  const int k0g = split * KSPLIT;

  for (int ch = 0; ch < NCHUNK; ++ch) {
    const int kbase = k0g + ch * CHUNK;
    // ---- stage 1: load & cvt X chunk, load key mask --------------------
    {
      const float* xs = mrna + ((size_t)b * LK + kbase) * DIM;
      for (int i = tid; i < CHUNK * DIM / 4; i += 256) {
        float4 v = reinterpret_cast<const float4*>(xs)[i];
        int o = i * 4;
        sX[o + 0] = (__bf16)v.x; sX[o + 1] = (__bf16)v.y;
        sX[o + 2] = (__bf16)v.z; sX[o + 3] = (__bf16)v.w;
      }
      if (tid < CHUNK) sMask[tid] = kmask[(size_t)b * LK + kbase + tid];
      if (ch + 1 < NCHUNK)   // global_prefetch_b8 next chunk (one line/thread)
        __builtin_prefetch(mrna + ((size_t)b * LK + kbase + CHUNK) * DIM + tid * 64, 0, 0);
    }
    __syncthreads();

    // ---- stage 2: Kc = X @ Wkv + bkv  (64x256, 8 tiles/wave) -----------
    #pragma unroll
    for (int i = 0; i < 8; ++i) {
      int t = wave * 8 + i;
      int m0 = (t & 3) * 16, n0 = (t >> 2) * 16;
      v8f c = {};
      #pragma unroll
      for (int cb = 0; cb < DIM; cb += 32) {
        v16bf a  = load_tile_A(sX, m0, DIM, cb, lane);
        v16bf bm = load_tile_B(WkvT, n0, DIM, cb, lane);
        c = wmma_bf16(a, bm, c);
      }
      int n = n0 + (lane & 15);
      float bias = bkv[n];
      #pragma unroll
      for (int r = 0; r < 8; ++r) {
        int m = m0 + r + hi8;
        __bf16 bv = (__bf16)(c[r] + bias);
        sK[m * DIM + n]   = bv;   // row-major  (B operand of Q@K^T)
        sKT[n * CHUNK + m] = bv;  // transposed (B operand of P@V)
      }
    }
    __syncthreads();

    // ---- stage 3: S = Q @ Kc^T  (64x64, 2 tiles/wave) ------------------
    #pragma unroll
    for (int i = 0; i < 2; ++i) {
      int t = wave * 2 + i;
      int m0 = (t & 3) * 16, n0 = (t >> 2) * 16;   // n over keys
      v8f c = {};
      #pragma unroll
      for (int cb = 0; cb < DIM; cb += 32) {
        v16bf a  = load_tile_A(sQ, m0, DIM, cb, lane);
        v16bf bm = load_tile_B(sK, n0, DIM, cb, lane);
        c = wmma_bf16(a, bm, c);
      }
      int n = n0 + (lane & 15);
      #pragma unroll
      for (int r = 0; r < 8; ++r) sS[(m0 + r + hi8) * CHUNK + n] = c[r];
    }
    __syncthreads();

    // ---- stage 4: online softmax row update (thread q per row) ---------
    if (tid < LQ) {
      float mold = sMrow[tid];
      float rmax = NEGBIG;
      for (int k = 0; k < CHUNK; ++k)
        if (sMask[k] != 0.f) rmax = fmaxf(rmax, sS[tid * CHUNK + k] * SCALE);
      float mnew = fmaxf(mold, rmax);
      float corr = __expf(mold - mnew);
      float l = sLrow[tid] * corr;
      for (int k = 0; k < CHUNK; ++k) {
        float p = 0.f;
        if (sMask[k] != 0.f) p = __expf(sS[tid * CHUNK + k] * SCALE - mnew);
        l += p;
        sP[tid * CHUNK + k] = (__bf16)p;
      }
      sMrow[tid] = mnew; sLrow[tid] = l; sCorr[tid] = corr;
    }
    __syncthreads();

    // ---- stage 5: acc = acc*corr + P @ Kc  (8 tiles/wave) --------------
    #pragma unroll
    for (int i = 0; i < 8; ++i) {
      int t = wave * 8 + i;
      int m0 = (t & 3) * 16, n0 = (t >> 2) * 16;   // n over D
      #pragma unroll
      for (int r = 0; r < 8; ++r) acc[i][r] *= sCorr[m0 + r + hi8];
      #pragma unroll
      for (int cb = 0; cb < CHUNK; cb += 32) {
        v16bf a  = load_tile_A(sP, m0, CHUNK, cb, lane);
        v16bf bm = load_tile_B(sKT, n0, CHUNK, cb, lane);
        acc[i] = wmma_bf16(a, bm, acc[i]);
      }
    }
    __syncthreads();
  }

  // ---- epilogue: spill partials -----------------------------------------
  const size_t sb = (size_t)b * NSPLIT + split;
  if (tid < LQ) {
    Mout[sb * LQ + tid] = sMrow[tid];
    Lout[sb * LQ + tid] = sLrow[tid];
  }
  #pragma unroll
  for (int i = 0; i < 8; ++i) {
    int t = wave * 8 + i;
    int m0 = (t & 3) * 16, n0 = (t >> 2) * 16;
    int n = n0 + (lane & 15);
    #pragma unroll
    for (int r = 0; r < 8; ++r)
      Acc[(sb * LQ + m0 + r + hi8) * DIM + n] = acc[i][r];
  }
}

// ---------------------------------------------------------------------------
// K3: combine split partials (log-sum-exp) + masked mean pool -> pooled[B,D]
// ---------------------------------------------------------------------------
__global__ __launch_bounds__(256) void k3_combine_pool(
    const float* __restrict__ Mout, const float* __restrict__ Lout,
    const float* __restrict__ Acc, const float* __restrict__ qmask,
    float* __restrict__ pooled) {
  __shared__ float sW[NSPLIT * LQ];
  __shared__ float sQm[LQ];
  __shared__ float sValid;
  const int b = blockIdx.x, tid = threadIdx.x;
  const size_t base = (size_t)b * NSPLIT;

  if (tid < LQ) {
    float mstar = NEGBIG;
    for (int s = 0; s < NSPLIT; ++s)
      mstar = fmaxf(mstar, Mout[(base + s) * LQ + tid]);
    float Lsum = 0.f;
    for (int s = 0; s < NSPLIT; ++s)
      Lsum += Lout[(base + s) * LQ + tid] * __expf(Mout[(base + s) * LQ + tid] - mstar);
    float inv = (Lsum > 0.f) ? 1.f / Lsum : 0.f;
    for (int s = 0; s < NSPLIT; ++s)
      sW[s * LQ + tid] = __expf(Mout[(base + s) * LQ + tid] - mstar) * inv;
    sQm[tid] = qmask[b * LQ + tid];
  }
  __syncthreads();
  if (tid == 0) {
    float v = 0.f;
    for (int q = 0; q < LQ; ++q) v += sQm[q];
    sValid = v;
  }
  __syncthreads();

  // thread = output channel d
  float out = 0.f;
  for (int q = 0; q < LQ; ++q) {
    float mq = sQm[q];
    if (mq != 0.f) {
      float inner = 0.f;
      for (int s = 0; s < NSPLIT; ++s) {
        float w = sW[s * LQ + q];
        if (w != 0.f)
          inner += w * Acc[((base + s) * LQ + q) * DIM + tid];
      }
      out += mq * inner;
    }
  }
  pooled[b * DIM + tid] = out / sValid;
}

// ---------------------------------------------------------------------------
// K4: MLP head: relu(pooled@W1+b1) -> relu(@W2+b2) -> @W3+b3  (tiny, VALU)
// ---------------------------------------------------------------------------
__global__ __launch_bounds__(512) void k4_mlp(
    const float* __restrict__ pooled,
    const float* __restrict__ W1, const float* __restrict__ b1,
    const float* __restrict__ W2, const float* __restrict__ b2,
    const float* __restrict__ W3, const float* __restrict__ b3,
    float* __restrict__ out) {
  __shared__ float sp[DIM];
  __shared__ float sh1[H1];
  __shared__ float sh2[H2];
  __shared__ float sred[512];
  const int b = blockIdx.x, tid = threadIdx.x;

  if (tid < DIM) sp[tid] = pooled[b * DIM + tid];
  __syncthreads();
  {
    float a = b1[tid];
    for (int c = 0; c < DIM; ++c) a += sp[c] * W1[c * H1 + tid];
    sh1[tid] = fmaxf(a, 0.f);
  }
  __syncthreads();
  {
    float a = b2[tid];
    for (int c = 0; c < H1; ++c) a += sh1[c] * W2[c * H2 + tid];
    sh2[tid] = fmaxf(a, 0.f);
  }
  __syncthreads();
  sred[tid] = sh2[tid] * W3[tid];
  __syncthreads();
  for (int s = 256; s > 0; s >>= 1) {
    if (tid < s) sred[tid] += sred[tid + s];
    __syncthreads();
  }
  if (tid == 0) out[b] = sred[0] + b3[0];
}

// ---------------------------------------------------------------------------
// launch
// ---------------------------------------------------------------------------
extern "C" void kernel_launch(void* const* d_in, const int* in_sizes, int n_in,
                              void* d_out, int out_size, void* d_ws, size_t ws_size,
                              hipStream_t stream) {
  const float* miRNA_hidden = (const float*)d_in[0];
  const float* mRNA_hidden  = (const float*)d_in[1];
  const float* miRNA_mask   = (const float*)d_in[2];
  const float* mRNA_mask    = (const float*)d_in[3];
  const float* Wq  = (const float*)d_in[4];
  const float* bq  = (const float*)d_in[5];
  const float* Wkv = (const float*)d_in[6];
  const float* bkv = (const float*)d_in[7];
  const float* W1  = (const float*)d_in[8];
  const float* b1  = (const float*)d_in[9];
  const float* W2  = (const float*)d_in[10];
  const float* b2  = (const float*)d_in[11];
  const float* W3  = (const float*)d_in[12];
  const float* b3  = (const float*)d_in[13];
  float* outp = (float*)d_out;

  char* ws = (char*)d_ws;
  size_t o = 0;
  __bf16* Qbf  = (__bf16*)(ws + o); o += (size_t)BATCH * LQ * DIM * 2;         // 1 MB
  __bf16* WqT  = (__bf16*)(ws + o); o += (size_t)DIM * DIM * 2;                // 128 KB
  __bf16* WkvT = (__bf16*)(ws + o); o += (size_t)DIM * DIM * 2;                // 128 KB
  float*  Mo   = (float*) (ws + o); o += (size_t)BATCH * NSPLIT * LQ * 4;      // 128 KB
  float*  Lo   = (float*) (ws + o); o += (size_t)BATCH * NSPLIT * LQ * 4;      // 128 KB
  float*  Acc  = (float*) (ws + o); o += (size_t)BATCH * NSPLIT * LQ * DIM * 4;// 32 MB
  float*  pooled = (float*)(ws + o); o += (size_t)BATCH * DIM * 4;             // 32 KB

  k0_convert_weights<<<dim3(2 * DIM * DIM / 256), dim3(256), 0, stream>>>(
      Wq, Wkv, WqT, WkvT);
  k1_qproj<<<dim3(BATCH), dim3(256), 0, stream>>>(miRNA_hidden, WqT, bq, Qbf);
  k2_flash_attn<<<dim3(BATCH * NSPLIT), dim3(256), SM_TOTAL, stream>>>(
      mRNA_hidden, mRNA_mask, Qbf, WkvT, bkv, Mo, Lo, Acc);
  k3_combine_pool<<<dim3(BATCH), dim3(256), 0, stream>>>(
      Mo, Lo, Acc, miRNA_mask, pooled);
  k4_mlp<<<dim3(BATCH), dim3(512), 0, stream>>>(
      pooled, W1, b1, W2, b2, W3, b3, outp);
}
